// TDLayer_43447889166904
// MI455X (gfx1250) — matrix operand order repaired and checked by reading, not compiled, not run on hardware
//
#include <hip/hip_runtime.h>

typedef __attribute__((ext_vector_type(16))) _Float16 v16h;
typedef __attribute__((ext_vector_type(8)))  float    v8f;
typedef __attribute__((ext_vector_type(4)))  float    v4f;

#define C_IN  32
#define C_OUT 64
#define KOFF  8
#define EPSV  1e-5f

// ---------------------------------------------------------------------------
// Kernel 1: zero the output accumulator and the stats buffer (sum, sumsq).
// ---------------------------------------------------------------------------
__global__ void zero_kernel(float* __restrict__ out, float* __restrict__ stats,
                            long total) {
  long idx    = (long)blockIdx.x * blockDim.x + threadIdx.x;
  long stride = (long)gridDim.x * blockDim.x;
  for (long i = idx; i < total; i += stride) out[i] = 0.0f;
  if (idx < 2 * C_OUT) stats[idx] = 0.0f;
}

// ---------------------------------------------------------------------------
// Kernel 2: gather -> masked WMMA GEMM -> atomic scatter.
// One wave (32 lanes) processes a tile of 16 input rows.
//   For j in 0..7:  A_j = rows with k_idx==j (others zeroed),  acc += A_j @ W[j]
// W is staged into LDS as f16, pre-swizzled to WMMA B-fragment layout.
// ---------------------------------------------------------------------------
__global__ __launch_bounds__(256) void conv_scatter_kernel(
    const float* __restrict__ x, const float* __restrict__ W,
    const int* __restrict__ k_idx, const int* __restrict__ out_idx,
    float* __restrict__ out, int n_rows, int n_tiles)
{
  // [j][t][lane][e] fragment-ready B layout: 8*4*32*16 halfs = 32 KB
  __shared__ _Float16 ldsB[KOFF * 4 * 32 * 16];

  const int tid = threadIdx.x;
  for (int idx = tid; idx < KOFF * 4 * 32 * 16; idx += 256) {
    int e = idx & 15;
    int L = (idx >> 4) & 31;
    int t = (idx >> 9) & 3;
    int j = idx >> 11;
    // B 32x16 f16 fragment as packed here (contiguous k, same order as A): lane L holds column N = L%16,
    // K = e for lanes 0-15, K = 16+e for lanes 16-31.
    int k   = ((L & 16) ? 16 : 0) + e;
    int col = t * 16 + (L & 15);
    ldsB[idx] = (_Float16)W[(j * C_IN + k) * C_OUT + col];
  }
  __syncthreads();

  const int wave = tid >> 5;
  const int lane = tid & 31;
  const int tile = blockIdx.x * 8 + wave;
  if (tile >= n_tiles) return;  // wave-uniform exit: WMMA waves keep EXEC=all-1s

  const int rowbase = tile * 16;
  const int r       = lane & 15;
  const int row     = rowbase + r;
  const bool rvalid = (row < n_rows);
  const int rclamp  = rvalid ? row : 0;
  const int kv      = rvalid ? k_idx[rclamp] : -1;

  // A 16x32 f16 fragment as packed here (contiguous k, same order as B): lane<16 holds row r with K in {0..15},
  // lane>=16 holds row r with K in {16..31}.
  const int kb = (lane < 16) ? 0 : 16;
  const float* xr = x + (size_t)rclamp * C_IN + kb;
  v4f a0 = *(const v4f*)(xr);
  v4f a1 = *(const v4f*)(xr + 4);
  v4f b0 = *(const v4f*)(xr + 8);
  v4f b1 = *(const v4f*)(xr + 12);
  v16h xv;
#pragma unroll
  for (int e = 0; e < 4; ++e) {
    xv[e]      = (_Float16)a0[e];
    xv[e + 4]  = (_Float16)a1[e];
    xv[e + 8]  = (_Float16)b0[e];
    xv[e + 12] = (_Float16)b1[e];
  }
  v16h zv = {};

  v8f acc[4] = {};
  const v16h* bfr = (const v16h*)ldsB;
#pragma unroll
  for (int j = 0; j < KOFF; ++j) {
    v16h av = (kv == j) ? xv : zv;  // mask rows not using kernel offset j
#pragma unroll
    for (int t = 0; t < 4; ++t) {
      v16h bv = bfr[(j * 4 + t) * 32 + lane];
      acc[t] = __builtin_amdgcn_wmma_f32_16x16x32_f16(
          false, av, false, bv, (short)0, acc[t], false, false);
    }
  }

  // Scatter: D layout => lane L, vgpr v holds (M = v + 8*(L>=16), N = L%16).
  const int hi = (lane >> 4) & 1;
  int oidx[8];
#pragma unroll
  for (int v = 0; v < 8; ++v) {
    int rr  = rowbase + hi * 8 + v;
    oidx[v] = (rr < n_rows) ? out_idx[rr] : -1;
  }
  const int col0 = lane & 15;
#pragma unroll
  for (int t = 0; t < 4; ++t) {
#pragma unroll
    for (int v = 0; v < 8; ++v) {
      if (oidx[v] >= 0)
        atomicAdd(&out[(size_t)oidx[v] * C_OUT + t * 16 + col0], acc[t][v]);
    }
  }
}

// ---------------------------------------------------------------------------
// Kernel 3: per-channel sum / sum-of-squares over (out + bias).
// threadIdx%64 = channel; block handles 4 rows per step, grid-strided.
// ---------------------------------------------------------------------------
__global__ __launch_bounds__(256) void stats_kernel(
    const float* __restrict__ out, const float* __restrict__ bias,
    float* __restrict__ stats, int num_out)
{
  const int c   = threadIdx.x & 63;
  const int grp = threadIdx.x >> 6;           // 0..3
  const float b = bias[c];
  float s = 0.0f, s2 = 0.0f;
  const int step = gridDim.x * 4;
  for (int rrow = blockIdx.x * 4 + grp; rrow < num_out; rrow += step) {
    float v = out[(size_t)rrow * C_OUT + c] + b;
    s  += v;
    s2 += v * v;
  }
  __shared__ float sh[512];
  sh[threadIdx.x]       = s;
  sh[256 + threadIdx.x] = s2;
  __syncthreads();
  if (threadIdx.x < 64) {
    float ts  = sh[threadIdx.x] + sh[threadIdx.x + 64] +
                sh[threadIdx.x + 128] + sh[threadIdx.x + 192];
    float ts2 = sh[256 + threadIdx.x] + sh[256 + threadIdx.x + 64] +
                sh[256 + threadIdx.x + 128] + sh[256 + threadIdx.x + 192];
    atomicAdd(&stats[threadIdx.x], ts);
    atomicAdd(&stats[64 + threadIdx.x], ts2);
  }
}

// ---------------------------------------------------------------------------
// Kernel 4: in-place BatchNorm (population stats) + ReLU.
// ---------------------------------------------------------------------------
__global__ void finalize_kernel(float* __restrict__ out,
                                const float* __restrict__ bias,
                                const float* __restrict__ gamma,
                                const float* __restrict__ beta,
                                const float* __restrict__ stats, int num_out)
{
  long total  = (long)num_out * C_OUT;
  long idx    = (long)blockIdx.x * blockDim.x + threadIdx.x;
  long stride = (long)gridDim.x * blockDim.x;
  const float inv = 1.0f / (float)num_out;
  for (long i = idx; i < total; i += stride) {
    int c      = (int)(i & 63);
    float mean = stats[c] * inv;
    float var  = stats[64 + c] * inv - mean * mean;
    float v    = out[i] + bias[c];
    v = (v - mean) * rsqrtf(var + EPSV) * gamma[c] + beta[c];
    out[i] = fmaxf(v, 0.0f);
  }
}

// ---------------------------------------------------------------------------
extern "C" void kernel_launch(void* const* d_in, const int* in_sizes, int n_in,
                              void* d_out, int out_size, void* d_ws, size_t ws_size,
                              hipStream_t stream) {
  const float* x      = (const float*)d_in[0];
  const float* W      = (const float*)d_in[1];
  const float* bias   = (const float*)d_in[2];
  const float* gamma  = (const float*)d_in[3];
  const float* beta   = (const float*)d_in[4];
  const int*   k_idx  = (const int*)d_in[5];
  const int*   out_idx= (const int*)d_in[6];
  (void)n_in; (void)ws_size;

  float* out   = (float*)d_out;
  float* stats = (float*)d_ws;        // 128 floats: [sum(64), sumsq(64)]

  const int n_rows  = in_sizes[0] / C_IN;
  const int num_out = out_size / C_OUT;
  const int n_tiles = (n_rows + 15) / 16;

  long total = (long)out_size;
  int zb = (int)((total + 255) / 256);
  if (zb > 2048) zb = 2048;
  zero_kernel<<<zb, 256, 0, stream>>>(out, stats, total);

  int cb = (n_tiles + 7) / 8;
  conv_scatter_kernel<<<cb, 256, 0, stream>>>(x, W, k_idx, out_idx, out,
                                              n_rows, n_tiles);

  stats_kernel<<<1024, 256, 0, stream>>>(out, bias, stats, num_out);

  int fb = (int)((total + 255) / 256);
  if (fb > 4096) fb = 4096;
  finalize_kernel<<<fb, 256, 0, stream>>>(out, bias, gamma, beta, stats, num_out);
}
